// BGHGNN_82952998355816
// MI455X (gfx1250) — matrix operand order repaired and checked by reading, not compiled, not run on hardware
//
#include <hip/hip_runtime.h>
#include <hip/hip_bf16.h>

// ---------------- problem constants (match reference) ----------------
#define N_NODES   100000
#define N_OUTK    50000
#define E_EDGES   1700000
#define RANK_     4
#define COMBINE_  64
#define IN_D      129     // IN_DIM + 1
#define EMB_D     513     // EMB + 1
#define EDG_D     513     // EDGE_DIM + 1
#define IN_P      160     // IN_D padded to 32
#define EMB_P     544     // EMB_D padded to 32
#define H0D       64      // HEADS*HIDDEN
#define H1D       128     // HEADS*OUT_DIM
#define SLOPE_    0.2f

typedef __attribute__((ext_vector_type(16))) _Float16 v16h;
typedef __attribute__((ext_vector_type(8)))  float    v8f;

// ---------------- WMMA helpers (gfx1250, wave32) ----------------
__device__ __forceinline__ v8f wmma_f16(v16h a, v16h b, v8f c) {
  // D = A(16x32) * B(32x16) + C, f32 accumulate
  return __builtin_amdgcn_wmma_f32_16x16x32_f16(false, a, false, b,
                                                (short)0, c, false, false);
}

// A fragment (16x32 f16) from row-major f32, no K mask (full step).
// ISA layout: lanes 0-15 -> M=lane, elems 0-7:K 0-7, elems 8-15:K 16-23;
// lanes 16-31 -> elems 0-7:K 8-15, elems 8-15:K 24-31.
__device__ __forceinline__ v16h load_a_fast(const float* __restrict__ A, int ld,
                                            int rowbase, int kbase, int lane) {
  const float* p = A + (long)(rowbase + (lane & 15)) * ld + kbase + ((lane >> 4) << 3);
  v16h a;
#pragma unroll
  for (int j = 0; j < 8; ++j) a[j] = (_Float16)p[j];
#pragma unroll
  for (int j = 0; j < 8; ++j) a[8 + j] = (_Float16)p[16 + j];
  return a;
}

// Masked A fragment (tail K step only).
__device__ __forceinline__ v16h load_a_mask(const float* __restrict__ A, int ld,
                                            int rowbase, int kbase, int K, int lane) {
  const int m = rowbase + (lane & 15);
  const int khalf = (lane >> 4) << 3;
  const float* row = A + (long)m * ld;
  v16h a;
#pragma unroll
  for (int j = 0; j < 16; ++j) {
    int k = kbase + khalf + ((j >> 3) << 4) + (j & 7);
    a[j] = (k < K) ? (_Float16)row[k] : (_Float16)0.0f;
  }
  return a;
}

// B fragment (32x16 f16, K x N) from pre-transposed f16 weights BT[n][Kpad]:
// lane holds column n = lane&15, elems 0-7: K = kbase + 8*(lane>>4) + j, elems 8-15: K = kbase + 16 + 8*(lane>>4) + j.
__device__ __forceinline__ v16h load_bt_frag(const _Float16* __restrict__ BT, int Kpad,
                                             int kbase, int colbase, int lane) {
  const int n = colbase + (lane & 15);
  const int k0 = kbase + ((lane >> 4) << 3);
  const _Float16* p = BT + (long)n * Kpad + k0;
  v16h b;
#pragma unroll
  for (int j = 0; j < 8; ++j) b[j] = p[j];
#pragma unroll
  for (int j = 0; j < 8; ++j) b[8 + j] = p[16 + j];
  return b;
}

// ---------------- small device helpers ----------------
__device__ __forceinline__ float leaky(float x) { return x > 0.f ? x : SLOPE_ * x; }

__device__ __forceinline__ void atomicMaxF(float* addr, float v) {
  if (v >= 0.f) atomicMax((int*)addr, __float_as_int(v));
  else          atomicMin((unsigned int*)addr, __float_as_uint(v));
}

// ---------------- kernels ----------------
__global__ void fill_kernel(float* __restrict__ p, float v, long n) {
  long i = (long)blockIdx.x * blockDim.x + threadIdx.x;
  if (i < n) p[i] = v;
}

// src: f32 [R][K][Ncols] row-major  ->  dst: f16 [R][Ncols][Kpad], zero-padded K.
__global__ void xpose_f16_kernel(const float* __restrict__ src, _Float16* __restrict__ dst,
                                 int K, int Ncols, int Kpad, long total) {
  long idx = (long)blockIdx.x * blockDim.x + threadIdx.x;
  if (idx >= total) return;
  int k = (int)(idx % Kpad);
  long rest = idx / Kpad;
  int n = (int)(rest % Ncols);
  long r = rest / Ncols;
  dst[idx] = (k < K) ? (_Float16)src[((long)r * K + k) * Ncols + n] : (_Float16)0.0f;
}

// CP tensor fusion: h[n,c] = relu(sum_r fw[r] * (hc@cf_r)*(ht@tf_r)*(hr@rf_r) + fb)
// block = 128 threads = 4 waves, wave r handles rank r; 16-node row tile per block.
__global__ void fusion_kernel(const float* __restrict__ hc, const float* __restrict__ ht,
                              const float* __restrict__ hr,
                              const _Float16* __restrict__ cfT,   // [R][64][IN_P]
                              const _Float16* __restrict__ tfT,   // [R][64][EMB_P]
                              const _Float16* __restrict__ rfT,   // [R][64][EMB_P]
                              const float* __restrict__ fw, const float* __restrict__ fb,
                              float* __restrict__ hout) {
  __shared__ float red[RANK_][16][COMBINE_];
  const int lane = threadIdx.x & 31;
  const int r = threadIdx.x >> 5;
  const int rowbase = blockIdx.x * 16;

  v8f ac[4] = {}, at[4] = {}, ag[4] = {};

  const _Float16* Bc = cfT + (long)r * COMBINE_ * IN_P;
  for (int ks = 0; ks < IN_P / 32; ++ks) {
    int kb = ks * 32;
    v16h a = (kb + 32 <= IN_D) ? load_a_fast(hc, IN_D, rowbase, kb, lane)
                               : load_a_mask(hc, IN_D, rowbase, kb, IN_D, lane);
    for (int ct = 0; ct < 4; ++ct)
      ac[ct] = wmma_f16(a, load_bt_frag(Bc, IN_P, kb, ct * 16, lane), ac[ct]);
  }
  const _Float16* Bt = tfT + (long)r * COMBINE_ * EMB_P;
  for (int ks = 0; ks < EMB_P / 32; ++ks) {
    int kb = ks * 32;
    v16h a = (kb + 32 <= EMB_D) ? load_a_fast(ht, EMB_D, rowbase, kb, lane)
                                : load_a_mask(ht, EMB_D, rowbase, kb, EMB_D, lane);
    for (int ct = 0; ct < 4; ++ct)
      at[ct] = wmma_f16(a, load_bt_frag(Bt, EMB_P, kb, ct * 16, lane), at[ct]);
  }
  const _Float16* Br = rfT + (long)r * COMBINE_ * EMB_P;
  for (int ks = 0; ks < EMB_P / 32; ++ks) {
    int kb = ks * 32;
    v16h a = (kb + 32 <= EDG_D) ? load_a_fast(hr, EDG_D, rowbase, kb, lane)
                                : load_a_mask(hr, EDG_D, rowbase, kb, EDG_D, lane);
    for (int ct = 0; ct < 4; ++ct)
      ag[ct] = wmma_f16(a, load_bt_frag(Br, EMB_P, kb, ct * 16, lane), ag[ct]);
  }

  const float w = fw[r];
  for (int ct = 0; ct < 4; ++ct)
#pragma unroll
    for (int j = 0; j < 8; ++j) {
      int row = j + ((lane >> 4) << 3);
      int col = ct * 16 + (lane & 15);
      red[r][row][col] = w * ac[ct][j] * at[ct][j] * ag[ct][j];
    }
  __syncthreads();
  for (int t = threadIdx.x; t < 16 * COMBINE_; t += blockDim.x) {
    int row = t >> 6, col = t & 63;
    float v = red[0][row][col] + red[1][row][col] + red[2][row][col] + red[3][row][col]
              + fb[col];
    hout[(long)(rowbase + row) * COMBINE_ + col] = fmaxf(v, 0.f);
  }
}

// Out[N x ncols] = A[N x 64] @ B[64 x ncols]; BT pre-transposed f16 [ncols][64].
// One wave per 16-col tile.
__global__ void gemm64_kernel(const float* __restrict__ A, const _Float16* __restrict__ BT,
                              float* __restrict__ Out, int ncols) {
  const int lane = threadIdx.x & 31;
  const int ct = threadIdx.x >> 5;
  const int rowbase = blockIdx.x * 16;
  v8f acc = {};
  for (int ks = 0; ks < 2; ++ks) {
    v16h a = load_a_fast(A, 64, rowbase, ks * 32, lane);
    v16h b = load_bt_frag(BT, 64, ks * 32, ct * 16, lane);
    acc = wmma_f16(a, b, acc);
  }
#pragma unroll
  for (int j = 0; j < 8; ++j) {
    int row = rowbase + j + ((lane >> 4) << 3);
    int col = ct * 16 + (lane & 15);
    Out[(long)row * ncols + col] = acc[j];
  }
}

// el[n,h] = sum_d z[n,h,d]*al[h,d]; er likewise. One wave per node.
__global__ void elr_kernel(const float* __restrict__ z, const float* __restrict__ al,
                           const float* __restrict__ ar, float* __restrict__ el,
                           float* __restrict__ er, int n, int dim) {
  const int node = (int)(((long)blockIdx.x * blockDim.x + threadIdx.x) >> 5);
  const int lane = threadIdx.x & 31;
  if (node >= n) return;
  const float* zr = z + (long)node * 2 * dim;
  for (int h = 0; h < 2; ++h) {
    float pl = 0.f, pr = 0.f;
    for (int d = lane; d < dim; d += 32) {
      float v = zr[h * dim + d];
      pl += v * al[h * dim + d];
      pr += v * ar[h * dim + d];
    }
    for (int off = 16; off > 0; off >>= 1) {
      pl += __shfl_xor(pl, off, 32);
      pr += __shfl_xor(pr, off, 32);
    }
    if (lane == 0) { el[node * 2 + h] = pl; er[node * 2 + h] = pr; }
  }
}

__global__ void edge_max_kernel(const float* __restrict__ el, const float* __restrict__ er,
                                const int* __restrict__ src, const int* __restrict__ dst,
                                float* __restrict__ m, long ne2) {
  long i = (long)blockIdx.x * blockDim.x + threadIdx.x;
  if (i >= ne2) return;
  int e = (int)(i >> 1), h = (int)(i & 1);
  int si = src[e], di = dst[e];
  float x = leaky(el[si * 2 + h] + er[di * 2 + h]);
  atomicMaxF(&m[di * 2 + h], x);
}

__global__ void edge_sum_kernel(const float* __restrict__ el, const float* __restrict__ er,
                                const float* __restrict__ m, const int* __restrict__ src,
                                const int* __restrict__ dst, float* __restrict__ s, long ne2) {
  long i = (long)blockIdx.x * blockDim.x + threadIdx.x;
  if (i >= ne2) return;
  int e = (int)(i >> 1), h = (int)(i & 1);
  int si = src[e], di = dst[e];
  float x = leaky(el[si * 2 + h] + er[di * 2 + h]);
  atomicAdd(&s[di * 2 + h], __expf(x - m[di * 2 + h]));
}

// alpha[e,h] = exp(e_eh - m[dst]) / s[dst]  -- one thread per (edge, head)
__global__ void alpha_kernel(const float* __restrict__ el, const float* __restrict__ er,
                             const float* __restrict__ m, const float* __restrict__ s,
                             const int* __restrict__ src, const int* __restrict__ dst,
                             float* __restrict__ alpha, long ne2) {
  long i = (long)blockIdx.x * blockDim.x + threadIdx.x;
  if (i >= ne2) return;
  int e = (int)(i >> 1), h = (int)(i & 1);
  int si = src[e], di = dst[e];
  float x = leaky(el[si * 2 + h] + er[di * 2 + h]);
  alpha[i] = __expf(x - m[di * 2 + h]) / s[di * 2 + h];
}

// layer-0 aggregation: one thread per (edge, d), d in [0,64)
__global__ void aggregate0_kernel(const float* __restrict__ z, const float* __restrict__ alpha,
                                  const int* __restrict__ src, const int* __restrict__ dst,
                                  float* __restrict__ rst, long total) {
  long idx = (long)blockIdx.x * blockDim.x + threadIdx.x;
  if (idx >= total) return;
  int e = (int)(idx >> 6), d = (int)(idx & 63), h = d >> 5;
  int si = src[e], di = dst[e];
  float a = alpha[(long)e * 2 + h];
  atomicAdd(&rst[(long)di * H0D + d], z[(long)si * H0D + d] * a);
}

// layer-1 aggregation (alpha precomputed in att); only dst < N_OUT contributes.
__global__ void aggregate1_kernel(const float* __restrict__ z, const float* __restrict__ alpha,
                                  const int* __restrict__ src, const int* __restrict__ dst,
                                  float* __restrict__ rst, long total) {
  long idx = (long)blockIdx.x * blockDim.x + threadIdx.x;
  if (idx >= total) return;
  int e = (int)(idx >> 7), d = (int)(idx & 127), h = d >> 6;
  int si = src[e], di = dst[e];
  if (di >= N_OUTK) return;
  float a = alpha[(long)e * 2 + h];
  atomicAdd(&rst[(long)di * H1D + d], z[(long)si * H1D + d] * a);
}

__global__ void bias_kernel(float* __restrict__ h, const float* __restrict__ b, long n) {
  long i = (long)blockIdx.x * blockDim.x + threadIdx.x;
  if (i < n) h[i] += b[i & 63];
}

// out[n,:] = rst1[n,:] + h1[n,:] @ Wres + b1  (rows < N_OUT); C-init from rst1+b1.
__global__ void finalize_kernel(const float* __restrict__ h1, const _Float16* __restrict__ WresT,
                                const float* __restrict__ b1, const float* __restrict__ rst1,
                                float* __restrict__ outh) {
  const int lane = threadIdx.x & 31;
  const int ct = threadIdx.x >> 5;          // 8 waves -> 128 cols
  const int rowbase = blockIdx.x * 16;
  v8f c;
#pragma unroll
  for (int j = 0; j < 8; ++j) {
    int row = rowbase + j + ((lane >> 4) << 3);
    int col = ct * 16 + (lane & 15);
    c[j] = rst1[(long)row * H1D + col] + b1[col];
  }
  for (int ks = 0; ks < 2; ++ks) {
    v16h a = load_a_fast(h1, H0D, rowbase, ks * 32, lane);
    v16h b = load_bt_frag(WresT, 64, ks * 32, ct * 16, lane);
    c = wmma_f16(a, b, c);
  }
#pragma unroll
  for (int j = 0; j < 8; ++j) {
    int row = rowbase + j + ((lane >> 4) << 3);
    int col = ct * 16 + (lane & 15);
    outh[(long)row * H1D + col] = c[j];
  }
}

// ---------------- host launcher ----------------
extern "C" void kernel_launch(void* const* d_in, const int* in_sizes, int n_in,
                              void* d_out, int out_size, void* d_ws, size_t ws_size,
                              hipStream_t stream) {
  (void)in_sizes; (void)n_in; (void)out_size; (void)ws_size;

  const float* hc   = (const float*)d_in[0];
  const float* ht   = (const float*)d_in[1];
  const float* hr   = (const float*)d_in[2];
  const float* cf   = (const float*)d_in[3];
  const float* tf   = (const float*)d_in[4];
  const float* rf   = (const float*)d_in[5];
  const float* fw   = (const float*)d_in[6];
  const float* fb   = (const float*)d_in[7];
  const float* W0   = (const float*)d_in[8];
  const float* al0  = (const float*)d_in[9];
  const float* ar0  = (const float*)d_in[10];
  const float* b0   = (const float*)d_in[11];
  const float* W1   = (const float*)d_in[12];
  const float* al1  = (const float*)d_in[13];
  const float* ar1  = (const float*)d_in[14];
  const float* b1   = (const float*)d_in[15];
  const float* Wres = (const float*)d_in[16];
  const int*   esrc = (const int*)d_in[17];
  const int*   edst = (const int*)d_in[18];

  float* out_h = (float*)d_out;                       // [50000 x 128]
  float* att   = out_h + (long)N_OUTK * H1D;          // [E x 2]

  // -------- workspace layout: f32 region then f16 region (32B aligned) --------
  float* ws   = (float*)d_ws;
  float* hf   = ws;                                   // N*64
  float* z0   = hf + (long)N_NODES * COMBINE_;        // N*64
  float* z1   = z0 + (long)N_NODES * H0D;             // N*128
  float* rst0 = z1 + (long)N_NODES * H1D;             // N*64   (zero)
  float* rst1 = rst0 + (long)N_NODES * H0D;           // NOUT*128 (zero)
  float* s0   = rst1 + (long)N_OUTK * H1D;            // N*2 (zero)
  float* s1   = s0 + (long)N_NODES * 2;               // N*2 (zero)
  float* m0   = s1 + (long)N_NODES * 2;               // N*2 (-inf)
  float* m1   = m0 + (long)N_NODES * 2;               // N*2 (-inf)
  float* el0  = m1 + (long)N_NODES * 2;
  float* er0  = el0 + (long)N_NODES * 2;
  float* el1  = er0 + (long)N_NODES * 2;
  float* er1  = el1 + (long)N_NODES * 2;
  float* alp0 = er1 + (long)N_NODES * 2;              // E*2

  _Float16* f16b  = (_Float16*)(alp0 + (long)E_EDGES * 2);
  _Float16* cfT   = f16b;                               // R*64*IN_P
  _Float16* tfT   = cfT  + (long)RANK_ * COMBINE_ * IN_P;
  _Float16* rfT   = tfT  + (long)RANK_ * COMBINE_ * EMB_P;
  _Float16* W0T   = rfT  + (long)RANK_ * COMBINE_ * EMB_P;   // 64*64
  _Float16* W1T   = W0T  + (long)H0D * 64;                    // 128*64
  _Float16* WresT = W1T  + (long)H1D * 64;                    // 128*64

  // -------- init --------
  const long nzero = (long)N_NODES * H0D + (long)N_OUTK * H1D + 2L * N_NODES * 2;
  fill_kernel<<<(int)((nzero + 255) / 256), 256, 0, stream>>>(rst0, 0.f, nzero);
  const long ninf = 2L * N_NODES * 2;
  fill_kernel<<<(int)((ninf + 255) / 256), 256, 0, stream>>>(m0, -__builtin_inff(), ninf);

  // -------- weight pre-transpose to f16 [n][Kpad] --------
  {
    long t;
    t = (long)RANK_ * COMBINE_ * IN_P;
    xpose_f16_kernel<<<(int)((t + 255) / 256), 256, 0, stream>>>(cf, cfT, IN_D, COMBINE_, IN_P, t);
    t = (long)RANK_ * COMBINE_ * EMB_P;
    xpose_f16_kernel<<<(int)((t + 255) / 256), 256, 0, stream>>>(tf, tfT, EMB_D, COMBINE_, EMB_P, t);
    xpose_f16_kernel<<<(int)((t + 255) / 256), 256, 0, stream>>>(rf, rfT, EDG_D, COMBINE_, EMB_P, t);
    t = (long)H0D * 64;
    xpose_f16_kernel<<<(int)((t + 255) / 256), 256, 0, stream>>>(W0, W0T, 64, H0D, 64, t);
    t = (long)H1D * 64;
    xpose_f16_kernel<<<(int)((t + 255) / 256), 256, 0, stream>>>(W1, W1T, 64, H1D, 64, t);
    xpose_f16_kernel<<<(int)((t + 255) / 256), 256, 0, stream>>>(Wres, WresT, 64, H1D, 64, t);
  }

  // -------- CP fusion -> hf [N,64] --------
  fusion_kernel<<<N_NODES / 16, 128, 0, stream>>>(hc, ht, hr, cfT, tfT, rfT, fw, fb, hf);

  // -------- GAT layer 0 --------
  gemm64_kernel<<<N_NODES / 16, 128, 0, stream>>>(hf, W0T, z0, H0D);
  elr_kernel<<<N_NODES / 4, 128, 0, stream>>>(z0, al0, ar0, el0, er0, N_NODES, 32);
  const long ne2 = 2L * E_EDGES;
  const int  gE  = (int)((ne2 + 255) / 256);
  edge_max_kernel<<<gE, 256, 0, stream>>>(el0, er0, esrc, edst, m0, ne2);
  edge_sum_kernel<<<gE, 256, 0, stream>>>(el0, er0, m0, esrc, edst, s0, ne2);
  alpha_kernel<<<gE, 256, 0, stream>>>(el0, er0, m0, s0, esrc, edst, alp0, ne2);
  const long agg0n = (long)E_EDGES * H0D;
  aggregate0_kernel<<<(int)((agg0n + 255) / 256), 256, 0, stream>>>(z0, alp0, esrc, edst,
                                                                    rst0, agg0n);
  const long nb0 = (long)N_NODES * H0D;
  bias_kernel<<<(int)((nb0 + 255) / 256), 256, 0, stream>>>(rst0, b0, nb0);  // rst0 -> h1

  // -------- GAT layer 1 --------
  gemm64_kernel<<<N_NODES / 16, 256, 0, stream>>>(rst0, W1T, z1, H1D);
  elr_kernel<<<N_NODES / 4, 128, 0, stream>>>(z1, al1, ar1, el1, er1, N_NODES, 64);
  edge_max_kernel<<<gE, 256, 0, stream>>>(el1, er1, esrc, edst, m1, ne2);
  edge_sum_kernel<<<gE, 256, 0, stream>>>(el1, er1, m1, esrc, edst, s1, ne2);
  alpha_kernel<<<gE, 256, 0, stream>>>(el1, er1, m1, s1, esrc, edst, att, ne2);  // att output
  const long agg1n = (long)E_EDGES * H1D;
  aggregate1_kernel<<<(int)((agg1n + 255) / 256), 256, 0, stream>>>(z1, att, esrc, edst,
                                                                    rst1, agg1n);

  // -------- residual GEMM + bias, rows < N_OUT --------
  finalize_kernel<<<N_OUTK / 16, 256, 0, stream>>>(rst0, WresT, b1, rst1, out_h);
}
